// Attention_7017976562403
// MI455X (gfx1250) — compile-verified, matrix operand order checked by reading, not run on hardware
//
#include <hip/hip_runtime.h>
#include <hip/hip_bf16.h>

#define DIM 768
#define HEADS 12
#define HD 64
#define BATCH 64
#define NTOK 256
#define SCALE 0.125f
#define LN_EPS 1e-5f

typedef _Float16 half16_t __attribute__((ext_vector_type(16)));
typedef _Float16 half4_t  __attribute__((ext_vector_type(4)));
typedef float    floatx8  __attribute__((ext_vector_type(8)));

// ---------------------------------------------------------------------------
// WMMA + async-copy helpers (gfx1250, wave32)
// ---------------------------------------------------------------------------
__device__ __forceinline__ floatx8 wmma_f16(half16_t a, half16_t b, floatx8 c) {
  return __builtin_amdgcn_wmma_f32_16x16x32_f16(false, a, false, b, (short)0, c,
                                                false, false);
}

// GLOBAL_LOAD_ASYNC_TO_LDS_B128 (GVS mode): LDS[vdst] = MEM[saddr + vaddr]
__device__ __forceinline__ void async_b128(unsigned lds_byte_addr, unsigned gl_byte_off,
                                           const void* base) {
  asm volatile("global_load_async_to_lds_b128 %0, %1, %2"
               :
               : "v"(lds_byte_addr), "v"(gl_byte_off),
                 "s"((unsigned long long)(uintptr_t)base)
               : "memory");
}
__device__ __forceinline__ void wait_async0() {
  asm volatile("s_wait_asynccnt 0x0" ::: "memory");
}
__device__ __forceinline__ void wait_async6() {
  asm volatile("s_wait_asynccnt 0x6" ::: "memory");
}
// low 32 bits of a generic LDS pointer == LDS byte offset (aperture low half is 0)
__device__ __forceinline__ unsigned lds_off(const void* p) {
  return (unsigned)(uintptr_t)p;
}

// A/B fragment gather from row-major [row][K]: element e of lane L holds
// K = kofs + ((L&16)?8:0) + e (e<8) / +16+e-8 (e>=8)  (ISA 16-bit A 16x32 layout)
__device__ __forceinline__ half16_t frag_ld(const _Float16* rowptr, int kofs, int lane) {
  half16_t f;
  int klo = kofs + ((lane & 16) ? 8 : 0);
#pragma unroll
  for (int e = 0; e < 8; ++e) f[e] = rowptr[klo + e];
#pragma unroll
  for (int e = 0; e < 8; ++e) f[8 + e] = rowptr[klo + 16 + e];
  return f;
}

// B fragment gather from [K][N] storage (V in P@V)
__device__ __forceinline__ half16_t frag_ld_col(const _Float16* base, int col, int ld,
                                                int kofs, int lane) {
  half16_t f;
  int klo = kofs + ((lane & 16) ? 8 : 0);
#pragma unroll
  for (int e = 0; e < 8; ++e) f[e] = base[(klo + e) * ld + col];
#pragma unroll
  for (int e = 0; e < 8; ++e) f[8 + e] = base[(klo + 16 + e) * ld + col];
  return f;
}

// ---------------------------------------------------------------------------
// small elementwise kernels
// ---------------------------------------------------------------------------
__global__ void modw_kernel(const float* __restrict__ w, const float* __restrict__ wmi,
                            const float* __restrict__ wmo, _Float16* __restrict__ out,
                            long total, int cols) {
  long idx = (long)blockIdx.x * 256 + threadIdx.x;
  if (idx >= total) return;
  int i = (int)(idx / cols);
  int j = (int)(idx % cols);
  float m = wmo[i * 2] * wmi[j * 2] + wmo[i * 2 + 1] * wmi[j * 2 + 1];
  out[idx] = (_Float16)(w[idx] * m);
}

__global__ void modb_kernel(const float* __restrict__ b, const float* __restrict__ bm,
                            const float* __restrict__ ba, float* __restrict__ out, int n) {
  int i = blockIdx.x * 256 + threadIdx.x;
  if (i < n) out[i] = b[i] * bm[i] + ba[i];
}

__global__ void cvt_f32_f16(const float* __restrict__ in, _Float16* __restrict__ out,
                            long n4) {
  long i = (long)blockIdx.x * 256 + threadIdx.x;
  if (i >= n4) return;
  float4 v = ((const float4*)in)[i];
  half4_t h;
  h[0] = (_Float16)v.x; h[1] = (_Float16)v.y;
  h[2] = (_Float16)v.z; h[3] = (_Float16)v.w;
  ((half4_t*)out)[i] = h;
}

// ---------------------------------------------------------------------------
// WMMA GEMM: C[M x Nfeat] = A(f16, MxK) * Bw(f16, Nfeat x K)^T + bias
// Block 128 thr = 4 waves (2x2), tile 64x128, wave tile 32x64 (8 wmma / K-step).
// K pipelined with double-buffered async global->LDS copies (ASYNCcnt).
// mode 0: scatter f16 into q/k/v [b][h][n][64];  mode 1: fp32 row-major Cout.
// ---------------------------------------------------------------------------
__global__ __launch_bounds__(128) void gemm_mod(const _Float16* __restrict__ A,
                                                const _Float16* __restrict__ Bw,
                                                const float* __restrict__ bias, int mode,
                                                _Float16* __restrict__ qb,
                                                _Float16* __restrict__ kb,
                                                _Float16* __restrict__ vb,
                                                float* __restrict__ Cout) {
  __shared__ _Float16 As[2][64][40];
  __shared__ _Float16 Bs[2][128][40];
  const int K = DIM;
  const int NSTEP = K / 32;
  int tid = threadIdx.x, lane = tid & 31, wave = tid >> 5;
  int wm = wave >> 1, wn = wave & 1;
  int m0 = blockIdx.y * 64, n0 = blockIdx.x * 128;
  int lane15 = lane & 15;
  unsigned asBase = lds_off(&As[0][0][0]);
  unsigned bsBase = lds_off(&Bs[0][0][0]);
  const unsigned asBufB = 64 * 40 * 2;
  const unsigned bsBufB = 128 * 40 * 2;

  // per-thread chunk coordinates (16B chunks, 8 halves)
  int arow0 = tid >> 2, ac8 = (tid & 3) * 8;          // A: 2 chunks (rows +0,+32)
  int brow0 = tid >> 2, bc8 = (tid & 3) * 8;          // B: 4 chunks (rows +0,32,64,96)

  auto issue_tiles = [&](int buf, int k0) {
#pragma unroll
    for (int it = 0; it < 2; ++it) {
      int row = arow0 + it * 32;
      unsigned lds = asBase + (unsigned)buf * asBufB + (unsigned)(row * 40 + ac8) * 2;
      unsigned go = (unsigned)(((m0 + row) * K + k0 + ac8) * 2);
      async_b128(lds, go, A);
    }
#pragma unroll
    for (int it = 0; it < 4; ++it) {
      int row = brow0 + it * 32;
      unsigned lds = bsBase + (unsigned)buf * bsBufB + (unsigned)(row * 40 + bc8) * 2;
      unsigned go = (unsigned)(((n0 + row) * K + k0 + bc8) * 2);
      async_b128(lds, go, Bw);
    }
  };

  floatx8 acc[2][4] = {};
  issue_tiles(0, 0);
  for (int s = 0; s < NSTEP; ++s) {
    int cur = s & 1;
    if (s + 1 < NSTEP) {
      issue_tiles(cur ^ 1, (s + 1) * 32);
      wait_async6();          // the 6 just-issued may remain outstanding
    } else {
      wait_async0();
    }
    __syncthreads();
    half16_t af[2], bf[4];
#pragma unroll
    for (int mi = 0; mi < 2; ++mi)
      af[mi] = frag_ld(&As[cur][wm * 32 + mi * 16 + lane15][0], 0, lane);
#pragma unroll
    for (int nj = 0; nj < 4; ++nj)
      bf[nj] = frag_ld(&Bs[cur][wn * 64 + nj * 16 + lane15][0], 0, lane);
#pragma unroll
    for (int mi = 0; mi < 2; ++mi)
#pragma unroll
      for (int nj = 0; nj < 4; ++nj) acc[mi][nj] = wmma_f16(af[mi], bf[nj], acc[mi][nj]);
    __syncthreads();
  }

  // ---- epilogue (all index math uniform except lane15/r) ----
  int rhalf = (lane & 16) ? 8 : 0;
  if (mode == 0) {
    int which = (n0 >= 2 * DIM) ? 2 : (n0 >= DIM ? 1 : 0);   // uniform: 768 % 128 == 0
    _Float16* dstbuf = (which == 0) ? qb : ((which == 1) ? kb : vb);
    int cc0 = n0 - which * DIM + wn * 64;
    int bi = m0 >> 8;
#pragma unroll
    for (int mi = 0; mi < 2; ++mi) {
      int nbase = (m0 & 255) + wm * 32 + mi * 16 + rhalf;
#pragma unroll
      for (int nj = 0; nj < 4; ++nj) {
        int h = (cc0 + nj * 16) >> 6;
        int d = nj * 16 + lane15;                             // (cc0 % 64) == 0
        float bv = bias[n0 + wn * 64 + nj * 16 + lane15];
        size_t base = (((size_t)(bi * HEADS + h) * NTOK) + nbase) * HD + d;
#pragma unroll
        for (int r = 0; r < 8; ++r)
          dstbuf[base + (size_t)r * HD] = (_Float16)(acc[mi][nj][r] + bv);
      }
    }
  } else {
#pragma unroll
    for (int mi = 0; mi < 2; ++mi) {
      int mrow = m0 + wm * 32 + mi * 16 + rhalf;
#pragma unroll
      for (int nj = 0; nj < 4; ++nj) {
        int c = n0 + wn * 64 + nj * 16 + lane15;
        float bv = bias[c];
#pragma unroll
        for (int r = 0; r < 8; ++r)
          Cout[(size_t)(mrow + r) * DIM + c] = acc[mi][nj][r] + bv;
      }
    }
  }
}

// ---------------------------------------------------------------------------
// RoPE + per-head LayerNorm (+ optional q scale), in place on f16 buffer.
// ---------------------------------------------------------------------------
__global__ __launch_bounds__(256) void rope_ln_kernel(_Float16* __restrict__ buf,
                                                      const float* __restrict__ sin_tab,
                                                      const float* __restrict__ cos_tab,
                                                      const float* __restrict__ g,
                                                      const float* __restrict__ bb,
                                                      float scale) {
  int lane = threadIdx.x & 31, wave = threadIdx.x >> 5;
  long rid = (long)blockIdx.x * 8 + wave;
  int n = (int)(rid % NTOK);
  int h = (int)((rid / NTOK) % HEADS);
  _Float16* p = buf + rid * HD;
  int d0 = lane * 2;
  float x0 = (float)p[d0], x1 = (float)p[d0 + 1];
  int cc = h * HD + d0;
  const float* cr = cos_tab + (size_t)n * DIM;
  const float* sr = sin_tab + (size_t)n * DIM;
  float y0 = x0 * cr[cc] + x1 * sr[cc];
  float y1 = x1 * cr[cc + 1] + x0 * sr[cc + 1];
  float sum = y0 + y1, sq = y0 * y0 + y1 * y1;
#pragma unroll
  for (int mask = 1; mask < 32; mask <<= 1) {
    sum += __shfl_xor(sum, mask, 32);
    sq += __shfl_xor(sq, mask, 32);
  }
  float mu = sum * (1.0f / HD);
  float var = sq * (1.0f / HD) - mu * mu;
  float rs = rsqrtf(var + LN_EPS);
  y0 = ((y0 - mu) * rs * g[d0] + bb[d0]) * scale;
  y1 = ((y1 - mu) * rs * g[d0 + 1] + bb[d0 + 1]) * scale;
  p[d0] = (_Float16)y0;
  p[d0 + 1] = (_Float16)y1;
}

// ---------------------------------------------------------------------------
// Fused flash attention per (b,h); K/V chunks async-copied into LDS.
// ---------------------------------------------------------------------------
__global__ __launch_bounds__(256) void attn_fused(const _Float16* __restrict__ qg,
                                                  const _Float16* __restrict__ kg,
                                                  const _Float16* __restrict__ vg,
                                                  _Float16* __restrict__ o16) {
  __shared__ _Float16 Kt[64][72];
  __shared__ _Float16 Vt[64][72];
  __shared__ _Float16 Pb[8][32][72];
  int tid = threadIdx.x, lane = tid & 31, wave = tid >> 5;
  int lane15 = lane & 15;
  int rhalf = (lane & 16) ? 8 : 0;
  int bh = blockIdx.x;
  int b = bh / HEADS, h = bh % HEADS;
  const _Float16* qptr = qg + (size_t)bh * NTOK * HD;
  const _Float16* kptr = kg + (size_t)bh * NTOK * HD;
  const _Float16* vptr = vg + (size_t)bh * NTOK * HD;
  unsigned ktBase = lds_off(&Kt[0][0]);
  unsigned vtBase = lds_off(&Vt[0][0]);

  half16_t qf[2][2];
#pragma unroll
  for (int mi = 0; mi < 2; ++mi)
#pragma unroll
    for (int ks = 0; ks < 2; ++ks)
      qf[mi][ks] =
          frag_ld(qptr + (size_t)(wave * 32 + mi * 16 + lane15) * HD, 32 * ks, lane);

  floatx8 oacc[2][4] = {};
  float rM[2][8], rL[2][8];
#pragma unroll
  for (int mi = 0; mi < 2; ++mi)
#pragma unroll
    for (int r = 0; r < 8; ++r) { rM[mi][r] = -1e30f; rL[mi][r] = 0.0f; }

  for (int c = 0; c < 4; ++c) {
#pragma unroll
    for (int it = 0; it < 2; ++it) {
      int slot = tid + it * 256;
      int row = slot >> 3;
      int c8 = (slot & 7) * 8;
      unsigned loff = (unsigned)(row * 72 + c8) * 2;
      unsigned go = (unsigned)(((c * 64 + row) * HD + c8) * 2);
      async_b128(ktBase + loff, go, kptr);
      async_b128(vtBase + loff, go, vptr);
    }
    wait_async0();
    __syncthreads();

    // S = q (32x64) @ Kchunk^T (64x64)
    floatx8 s[2][4] = {};
#pragma unroll
    for (int nj = 0; nj < 4; ++nj)
#pragma unroll
      for (int ks = 0; ks < 2; ++ks) {
        half16_t bf = frag_ld(&Kt[nj * 16 + lane15][0], 32 * ks, lane);
        s[0][nj] = wmma_f16(qf[0][ks], bf, s[0][nj]);
        s[1][nj] = wmma_f16(qf[1][ks], bf, s[1][nj]);
      }

    // online softmax (row stats live in 16-lane halves of the D layout)
#pragma unroll
    for (int mi = 0; mi < 2; ++mi)
#pragma unroll
      for (int r = 0; r < 8; ++r) {
        float mx = fmaxf(fmaxf(s[mi][0][r], s[mi][1][r]), fmaxf(s[mi][2][r], s[mi][3][r]));
        mx = fmaxf(mx, __shfl_xor(mx, 1, 32));
        mx = fmaxf(mx, __shfl_xor(mx, 2, 32));
        mx = fmaxf(mx, __shfl_xor(mx, 4, 32));
        mx = fmaxf(mx, __shfl_xor(mx, 8, 32));
        float nM = fmaxf(rM[mi][r], mx);
        float sc = __expf(rM[mi][r] - nM);
        float rsum = 0.0f;
#pragma unroll
        for (int nj = 0; nj < 4; ++nj) {
          float pv = __expf(s[mi][nj][r] - nM);
          s[mi][nj][r] = pv;
          rsum += pv;
        }
        rsum += __shfl_xor(rsum, 1, 32);
        rsum += __shfl_xor(rsum, 2, 32);
        rsum += __shfl_xor(rsum, 4, 32);
        rsum += __shfl_xor(rsum, 8, 32);
        rL[mi][r] = rL[mi][r] * sc + rsum;
        rM[mi][r] = nM;
#pragma unroll
        for (int dj = 0; dj < 4; ++dj) oacc[mi][dj][r] *= sc;
      }

    // restage P (D-layout -> row-major) in wave-private LDS
#pragma unroll
    for (int mi = 0; mi < 2; ++mi)
#pragma unroll
      for (int nj = 0; nj < 4; ++nj)
#pragma unroll
        for (int r = 0; r < 8; ++r)
          Pb[wave][mi * 16 + rhalf + r][nj * 16 + lane15] = (_Float16)s[mi][nj][r];

    // O += P (32x64) @ Vchunk (64x64)
#pragma unroll
    for (int ks = 0; ks < 2; ++ks) {
      half16_t ap0 = frag_ld(&Pb[wave][lane15][0], 32 * ks, lane);
      half16_t ap1 = frag_ld(&Pb[wave][16 + lane15][0], 32 * ks, lane);
#pragma unroll
      for (int dj = 0; dj < 4; ++dj) {
        half16_t bv = frag_ld_col(&Vt[0][0], dj * 16 + lane15, 72, 32 * ks, lane);
        oacc[0][dj] = wmma_f16(ap0, bv, oacc[0][dj]);
        oacc[1][dj] = wmma_f16(ap1, bv, oacc[1][dj]);
      }
    }
    __syncthreads();
  }

  // normalize; write o as f16 [b][n][h*64+d] for the proj GEMM
#pragma unroll
  for (int mi = 0; mi < 2; ++mi)
#pragma unroll
    for (int r = 0; r < 8; ++r) {
      float inv = 1.0f / rL[mi][r];
      int m = wave * 32 + mi * 16 + rhalf + r;
      size_t row = (size_t)(b * NTOK + m);
#pragma unroll
      for (int dj = 0; dj < 4; ++dj) {
        int col = h * HD + dj * 16 + lane15;
        o16[row * DIM + col] = (_Float16)(oacc[mi][dj][r] * inv);
      }
    }
}

// ---------------------------------------------------------------------------
extern "C" void kernel_launch(void* const* d_in, const int* in_sizes, int n_in,
                              void* d_out, int out_size, void* d_ws, size_t ws_size,
                              hipStream_t stream) {
  (void)in_sizes; (void)n_in; (void)out_size; (void)ws_size;
  const float* x        = (const float*)d_in[0];
  const float* qkv_w    = (const float*)d_in[1];
  const float* qkv_b    = (const float*)d_in[2];
  const float* qkv_wmi  = (const float*)d_in[3];
  const float* qkv_wmo  = (const float*)d_in[4];
  const float* qkv_bm   = (const float*)d_in[5];
  const float* qkv_ba   = (const float*)d_in[6];
  const float* proj_w   = (const float*)d_in[7];
  const float* proj_b   = (const float*)d_in[8];
  const float* proj_wmi = (const float*)d_in[9];
  const float* proj_wmo = (const float*)d_in[10];
  const float* proj_bm  = (const float*)d_in[11];
  const float* proj_ba  = (const float*)d_in[12];
  const float* qn_w     = (const float*)d_in[13];
  const float* qn_b     = (const float*)d_in[14];
  const float* kn_w     = (const float*)d_in[15];
  const float* kn_b     = (const float*)d_in[16];
  const float* sin_tab  = (const float*)d_in[17];
  const float* cos_tab  = (const float*)d_in[18];
  float* out = (float*)d_out;

  const size_t Mrows = (size_t)BATCH * NTOK;                    // 16384
  const size_t qkvfeat = 3 * DIM;                               // 2304
  const size_t headElems = (size_t)BATCH * HEADS * NTOK * HD;   // 12,582,912

  char* ws = (char*)d_ws;
  size_t off = 0;
  auto take = [&](size_t bytes) -> void* {
    size_t cur = (off + 255) & ~(size_t)255;
    off = cur + bytes;
    return (void*)(ws + cur);
  };
  _Float16* x16  = (_Float16*)take(Mrows * DIM * 2);
  _Float16* wq16 = (_Float16*)take(qkvfeat * DIM * 2);
  _Float16* wp16 = (_Float16*)take((size_t)DIM * DIM * 2);
  float*    bq   = (float*)take(qkvfeat * 4);
  float*    bp   = (float*)take(DIM * 4);
  _Float16* qb   = (_Float16*)take(headElems * 2);
  _Float16* kb   = (_Float16*)take(headElems * 2);
  _Float16* vb   = (_Float16*)take(headElems * 2);
  _Float16* o16  = (_Float16*)take(headElems * 2);

  // 0) x -> f16
  {
    long n4 = (long)(Mrows * DIM / 4);
    cvt_f32_f16<<<(unsigned)((n4 + 255) / 256), 256, 0, stream>>>(x, x16, n4);
  }
  // 1) modulated weights/biases
  {
    long t1 = (long)qkvfeat * DIM;
    modw_kernel<<<(unsigned)((t1 + 255) / 256), 256, 0, stream>>>(qkv_w, qkv_wmi, qkv_wmo,
                                                                  wq16, t1, DIM);
    long t2 = (long)DIM * DIM;
    modw_kernel<<<(unsigned)((t2 + 255) / 256), 256, 0, stream>>>(proj_w, proj_wmi,
                                                                  proj_wmo, wp16, t2, DIM);
    modb_kernel<<<(unsigned)((qkvfeat + 255) / 256), 256, 0, stream>>>(qkv_b, qkv_bm,
                                                                       qkv_ba, bq,
                                                                       (int)qkvfeat);
    modb_kernel<<<(unsigned)((DIM + 255) / 256), 256, 0, stream>>>(proj_b, proj_bm,
                                                                   proj_ba, bp, DIM);
  }
  // 2) QKV GEMM -> q/k/v f16 [b][h][n][64]
  {
    dim3 grid((unsigned)(qkvfeat / 128), (unsigned)(Mrows / 64));
    gemm_mod<<<grid, 128, 0, stream>>>(x16, wq16, bq, 0, qb, kb, vb, nullptr);
  }
  // 3) RoPE + LN (+scale for q)
  {
    unsigned nblk = (unsigned)((size_t)BATCH * HEADS * NTOK / 8);
    rope_ln_kernel<<<nblk, 256, 0, stream>>>(qb, sin_tab, cos_tab, qn_w, qn_b, SCALE);
    rope_ln_kernel<<<nblk, 256, 0, stream>>>(kb, sin_tab, cos_tab, kn_w, kn_b, 1.0f);
  }
  // 4) fused attention per (b,h) -> o16
  attn_fused<<<BATCH * HEADS, 256, 0, stream>>>(qb, kb, vb, o16);
  // 5) projection GEMM -> d_out (fp32)
  {
    dim3 grid((unsigned)(DIM / 128), (unsigned)(Mrows / 64));
    gemm_mod<<<grid, 128, 0, stream>>>(o16, wp16, bp, 1, nullptr, nullptr, nullptr, out);
  }
}
